// Convolution_48223892799999
// MI455X (gfx1250) — matrix operand order repaired and checked by reading, not compiled, not run on hardware
//
#include <hip/hip_runtime.h>
#include <hip/hip_bf16.h>

// ---------------- problem constants ----------------
#define N_NODES 10000
#define N_EDGES 160000
// MUL0=128, MUL1=64, FC_IN=8, FC_H=64, MID=192, agg width = 768

typedef __attribute__((ext_vector_type(16))) _Float16 v16h;
typedef __attribute__((ext_vector_type(8)))  _Float16 v8h;
typedef __attribute__((ext_vector_type(8)))  float    v8f;

#define INV8   0.35355339059327373f                 // 1/sqrt(8)
#define INV128 0.08838834764831845f                 // 1/sqrt(128)
#define INV64S 0.125f                               // 1/sqrt(64)
#define INV192 0.07216878364870323f                 // 1/sqrt(192)
#define INV3   0.5773502691896258f                  // 1/sqrt(3)
#define COS_A  0.9238795325112867f                  // cos(pi/8)
#define SIN_A  0.3826834323650898f                  // sin(pi/8)

// ---------------- device helpers ----------------
__device__ __forceinline__ float gelu(float x) {
  // jax.nn.gelu default (tanh approximation)
  float x3 = x * x * x;
  float t  = tanhf(0.7978845608028654f * (x + 0.044715f * x3));
  return 0.5f * x * (1.0f + t);
}

// A fragment (16x32 f16, row-major source, leading dim = lda):
// lanes 0-15: row m=lane, K = kbase+{0..7} and kbase+{16..23}
// lanes 16-31: row m=lane-16, K = kbase+{8..15} and kbase+{24..31}
__device__ __forceinline__ v16h wmma_load_a(const _Float16* A, int lda, int m0,
                                            int kbase, int lane) {
  int m  = m0 + (lane & 15);
  int kb = kbase + ((lane >> 4) << 3);
  const _Float16* p = A + (size_t)m * lda + kb;
  v8h lo = *(const v8h*)(p);
  v8h hi = *(const v8h*)(p + 16);
  return __builtin_shufflevector(lo, hi, 0,1,2,3,4,5,6,7,8,9,10,11,12,13,14,15);
}

// B fragment (32x16 f16) from a TRANSPOSED weight BT[N][K] (leading dim = ldb=K):
// lane l<16: column n=n0+l, K = kbase+{0..7} and kbase+{16..23} ; lane>=16: K = kbase+{8..15} and kbase+{24..31}
__device__ __forceinline__ v16h wmma_load_b(const _Float16* BT, int ldb, int n0,
                                            int kbase, int lane) {
  int n  = n0 + (lane & 15);
  int kk = kbase + ((lane >> 4) << 3);
  const _Float16* p = BT + (size_t)n * ldb + kk;
  v8h lo = *(const v8h*)(p);
  v8h hi = *(const v8h*)(p + 16);
  return __builtin_shufflevector(lo, hi, 0,1,2,3,4,5,6,7,8,9,10,11,12,13,14,15);
}

// A fragment assembled from an f32 LDS tile (row stride 64)
__device__ __forceinline__ v16h lds_load_a(const float* H, int kbase, int lane) {
  int m  = lane & 15;
  int kb = kbase + ((lane >> 4) << 3);
  const float* p = H + m * 64 + kb;
  v16h a;
#pragma unroll
  for (int j = 0; j < 8; ++j) {
    a[j]     = (_Float16)p[j];
    a[8 + j] = (_Float16)p[16 + j];
  }
  return a;
}

__device__ __forceinline__ v8f wmma_f16(v16h a, v16h b, v8f c) {
  return __builtin_amdgcn_wmma_f32_16x16x32_f16(false, a, false, b, (short)0, c,
                                                false, false);
}

// ---------------- prep kernels ----------------
// transpose + f32->f16 convert:  WT[n*K + k] = W[k*N + n]
__global__ void k_w2h(const float* __restrict__ W, _Float16* __restrict__ WT,
                      int K, int N) {
  int i = blockIdx.x * 256 + threadIdx.x;
  if (i >= K * N) return;
  int k = i / N, n = i % N;
  WT[n * K + k] = (_Float16)W[i];
}

// node_input -> xa0[n][128] (f16) and xa1[(n*3+i)][64] (f16)
__global__ void k_x(const float* __restrict__ X, _Float16* __restrict__ xa0,
                    _Float16* __restrict__ xa1) {
  int i = blockIdx.x * 256 + threadIdx.x;
  if (i >= N_NODES * 320) return;
  int n = i / 320, c = i % 320;
  float v = X[i];
  if (c < 128) {
    xa0[n * 128 + c] = (_Float16)v;
  } else {
    int cc = c - 128;
    int vch = cc / 3, ii = cc % 3;
    xa1[((size_t)n * 3 + ii) * 64 + vch] = (_Float16)v;
  }
}

__global__ void k_zero(float* __restrict__ p, int count) {
  int i = blockIdx.x * 256 + threadIdx.x;
  if (i < count) p[i] = 0.0f;
}

// agg -> za0[n][192] and za1[(n*3+i)][192] (f16 A-buffers for output GEMMs)
__global__ void k_repack(const float* __restrict__ agg, _Float16* __restrict__ za0,
                         _Float16* __restrict__ za1) {
  int i = blockIdx.x * 256 + threadIdx.x;
  if (i >= N_NODES * 768) return;
  int n = i / 768, c = i % 768;
  float v = agg[i];
  if (c < 192) {
    za0[(size_t)n * 192 + c] = (_Float16)v;
  } else {
    int cc = c - 192;
    int vv = cc / 3, ii = cc % 3;
    za1[((size_t)n * 3 + ii) * 192 + vv] = (_Float16)v;
  }
}

// ---------------- WMMA GEMM: two outputs sharing A (feat & self) ----------------
__global__ __launch_bounds__(256) void k_gemm_dual(
    const _Float16* __restrict__ A, const _Float16* __restrict__ B1T,
    const _Float16* __restrict__ B2T, float* __restrict__ O1,
    float* __restrict__ O2, int n_mtiles, int n_ntiles, int K, int N,
    float scale) {
  int lane = threadIdx.x & 31;
  int gw   = blockIdx.x * 8 + (threadIdx.x >> 5);
  if (gw >= n_mtiles * n_ntiles) return;
  int mt = gw / n_ntiles, nt = gw % n_ntiles;
  int m0 = mt * 16, n0 = nt * 16;
  v8f acc1 = {}, acc2 = {};
  for (int k = 0; k < K; k += 32) {
    v16h a = wmma_load_a(A, K, m0, k, lane);
    acc1 = wmma_f16(a, wmma_load_b(B1T, K, n0, k, lane), acc1);
    acc2 = wmma_f16(a, wmma_load_b(B2T, K, n0, k, lane), acc2);
  }
  int n  = n0 + (lane & 15);
  int mb = m0 + ((lane >> 4) << 3);
#pragma unroll
  for (int r = 0; r < 8; ++r) {
    O1[(size_t)(mb + r) * N + n] = acc1[r] * scale;
    O2[(size_t)(mb + r) * N + n] = acc2[r] * scale;
  }
}

// ---------------- fused edge kernel ----------------
// one wave = 16 edges: scalar layer1 (K=8), WMMA layer2 (64x64),
// WMMA heads (64x384) fused with elementwise tensor product + atomic scatter.
// w0/w1 heads share the feat0 gathers; w2/w3 heads share the feat1 gathers.
__global__ __launch_bounds__(256) void k_edge(
    const float* __restrict__ esa, const float* __restrict__ edge_attr,
    const int* __restrict__ edge_src, const int* __restrict__ edge_dst,
    const float* __restrict__ M1, const _Float16* __restrict__ M2T,
    const _Float16* __restrict__ Wtp0T, const _Float16* __restrict__ Wtp1T,
    const _Float16* __restrict__ Wtp2T, const _Float16* __restrict__ Wtp3T,
    const float* __restrict__ feat0, const float* __restrict__ feat1,
    float* __restrict__ agg) {
  __shared__ float Hs[8][16 * 64];   // per-wave 16x64 activation tile
  __shared__ float Es[8][16 * 8];    // per-wave edge metadata
  int lane  = threadIdx.x & 31;
  int wv    = threadIdx.x >> 5;
  int ebase = (blockIdx.x * 8 + wv) * 16;   // 160000/16 = 10000 waves, exact
  float* H  = Hs[wv];
  float* MT = Es[wv];

  if (lane < 16) {
    int e = ebase + lane;
    int s = edge_src[e];
    ((int*)MT)[lane * 8 + 0] = s;
    ((int*)MT)[lane * 8 + 1] = edge_dst[e];
    MT[lane * 8 + 2] = edge_attr[e * 4 + 0];
    MT[lane * 8 + 3] = edge_attr[e * 4 + 1];
    MT[lane * 8 + 4] = edge_attr[e * 4 + 2];
    MT[lane * 8 + 5] = edge_attr[e * 4 + 3];
    // prefetch the gathered rows (consumed after two WMMA stages);
    // lowers to global_prefetch_b8, one per 128B line
    const float* p0 = feat0 + (size_t)s * 128;   // 512 B
    const float* p1 = feat1 + (size_t)s * 3 * 64; // 768 B
#pragma unroll
    for (int q = 0; q < 4; ++q) __builtin_prefetch(p0 + q * 32, 0, 3);
#pragma unroll
    for (int q = 0; q < 6; ++q) __builtin_prefetch(p1 + q * 32, 0, 3);
  }

  // layer 1: t1 = gelu(esa @ M1 / sqrt(8))   (K=8: scalar)
  for (int idx = lane; idx < 16 * 64; idx += 32) {
    int r = idx >> 6, c = idx & 63;
    const float* es = esa + (size_t)(ebase + r) * 8;
    float s = 0.0f;
#pragma unroll
    for (int j = 0; j < 8; ++j) s += es[j] * M1[j * 64 + c];
    H[r * 64 + c] = gelu(s * INV8);
  }
  __syncthreads();

  // layer 2: h = gelu(t1 @ M2 / sqrt(64))  via WMMA, round-trip through LDS
  v16h a0 = lds_load_a(H, 0, lane);
  v16h a1 = lds_load_a(H, 32, lane);
#pragma unroll
  for (int nt = 0; nt < 4; ++nt) {
    v8f acc = {};
    acc = wmma_f16(a0, wmma_load_b(M2T, 64, nt * 16, 0, lane), acc);
    acc = wmma_f16(a1, wmma_load_b(M2T, 64, nt * 16, 32, lane), acc);
    int n  = nt * 16 + (lane & 15);
    int mb = (lane >> 4) << 3;
#pragma unroll
    for (int r = 0; r < 8; ++r) H[(mb + r) * 64 + n] = gelu(acc[r] * INV64S);
  }
  __syncthreads();

  v16h h0 = lds_load_a(H, 0, lane);
  v16h h1 = lds_load_a(H, 32, lane);

  // per-lane edge metadata for the 8 rows this lane owns in D fragments
  int mb = (lane >> 4) << 3;
  int esrc[8], edst[8];
  float ey0[8], y1x[8], y1y[8], y1z[8];
#pragma unroll
  for (int r = 0; r < 8; ++r) {
    esrc[r] = ((int*)MT)[(mb + r) * 8 + 0];
    edst[r] = ((int*)MT)[(mb + r) * 8 + 1];
    ey0[r]  = MT[(mb + r) * 8 + 2];
    y1x[r]  = MT[(mb + r) * 8 + 3];
    y1y[r]  = MT[(mb + r) * 8 + 4];
    y1z[r]  = MT[(mb + r) * 8 + 5];
  }

  // heads w0 & w1 (c in 0..127): one feat0 gather feeds 4 atomics
  for (int t = 0; t < 8; ++t) {
    int nloc = t * 16;
    v8f acc0 = {}, acc1 = {};
    acc0 = wmma_f16(h0, wmma_load_b(Wtp0T, 64, nloc, 0, lane), acc0);
    acc0 = wmma_f16(h1, wmma_load_b(Wtp0T, 64, nloc, 32, lane), acc0);
    acc1 = wmma_f16(h0, wmma_load_b(Wtp1T, 64, nloc, 0, lane), acc1);
    acc1 = wmma_f16(h1, wmma_load_b(Wtp1T, 64, nloc, 32, lane), acc1);
    int c = nloc + (lane & 15);
#pragma unroll
    for (int r = 0; r < 8; ++r) {
      float e0v = feat0[(size_t)esrc[r] * 128 + c];
      float* ad = agg + (size_t)edst[r] * 768;
      float w0  = acc0[r] * INV64S;              // w0 = h @ Wtp0 / sqrt(64)
      float w1e = acc1[r] * INV64S * e0v;        // w1 * e0
      atomicAdd(ad + c, w0 * e0v * ey0[r]);                    // mid0a
      atomicAdd(ad + 192 + c * 3 + 0, w1e * y1x[r]);           // mid1a
      atomicAdd(ad + 192 + c * 3 + 1, w1e * y1y[r]);
      atomicAdd(ad + 192 + c * 3 + 2, w1e * y1z[r]);
    }
  }

  // heads w2 & w3 (c in 0..63): one feat1 (3-vector) gather feeds 4 atomics
  for (int t = 0; t < 4; ++t) {
    int nloc = t * 16;
    v8f acc2 = {}, acc3 = {};
    acc2 = wmma_f16(h0, wmma_load_b(Wtp2T, 64, nloc, 0, lane), acc2);
    acc2 = wmma_f16(h1, wmma_load_b(Wtp2T, 64, nloc, 32, lane), acc2);
    acc3 = wmma_f16(h0, wmma_load_b(Wtp3T, 64, nloc, 0, lane), acc3);
    acc3 = wmma_f16(h1, wmma_load_b(Wtp3T, 64, nloc, 32, lane), acc3);
    int c = nloc + (lane & 15);
#pragma unroll
    for (int r = 0; r < 8; ++r) {
      int src   = esrc[r];
      float e1a = feat1[((size_t)src * 3 + 0) * 64 + c];
      float e1b = feat1[((size_t)src * 3 + 1) * 64 + c];
      float e1c = feat1[((size_t)src * 3 + 2) * 64 + c];
      float* ad = agg + (size_t)edst[r] * 768;
      float wy  = acc2[r] * INV64S * ey0[r];     // w2 * y0
      atomicAdd(ad + 576 + c * 3 + 0, wy * e1a);               // mid1b
      atomicAdd(ad + 576 + c * 3 + 1, wy * e1b);
      atomicAdd(ad + 576 + c * 3 + 2, wy * e1c);
      float s3 = e1a * y1x[r] + e1b * y1y[r] + e1c * y1z[r];
      atomicAdd(ad + 128 + c, acc3[r] * INV64S * s3 * INV3);   // mid0b
    }
  }
}

// ---------------- output GEMM + skip combine ----------------
__global__ __launch_bounds__(256) void k_out0(const _Float16* __restrict__ za0,
                                              const _Float16* __restrict__ Wo0T,
                                              const float* __restrict__ self0,
                                              float* __restrict__ out) {
  int lane = threadIdx.x & 31;
  int gw   = blockIdx.x * 8 + (threadIdx.x >> 5);  // 625*8 = 5000 waves exact
  int mt = gw >> 3, nt = gw & 7;
  int m0 = mt * 16, n0 = nt * 16;
  v8f acc = {};
  for (int k = 0; k < 192; k += 32) {
    v16h a = wmma_load_a(za0, 192, m0, k, lane);
    acc = wmma_f16(a, wmma_load_b(Wo0T, 192, n0, k, lane), acc);
  }
  int n  = n0 + (lane & 15);
  int mb = m0 + ((lane >> 4) << 3);
  const float sc = SIN_A * 0.25f * INV192;  // fold 1/sqrt(16) and 1/sqrt(192)
#pragma unroll
  for (int r = 0; r < 8; ++r) {
    int m = mb + r;
    out[(size_t)m * 320 + n] = COS_A * self0[(size_t)m * 128 + n] + sc * acc[r];
  }
}

__global__ __launch_bounds__(256) void k_out1(const _Float16* __restrict__ za1,
                                              const _Float16* __restrict__ Wo1T,
                                              const float* __restrict__ self1,
                                              float* __restrict__ out) {
  int lane = threadIdx.x & 31;
  int gw   = blockIdx.x * 8 + (threadIdx.x >> 5);
  if (gw >= 1875 * 4) return;  // 30000 rows -> 1875 m-tiles, 4 n-tiles
  int mt = gw >> 2, nt = gw & 3;
  int m0 = mt * 16, n0 = nt * 16;
  v8f acc = {};
  for (int k = 0; k < 192; k += 32) {
    v16h a = wmma_load_a(za1, 192, m0, k, lane);
    acc = wmma_f16(a, wmma_load_b(Wo1T, 192, n0, k, lane), acc);
  }
  int u  = n0 + (lane & 15);
  int mb = m0 + ((lane >> 4) << 3);
  const float sc = SIN_A * 0.25f * INV192;
#pragma unroll
  for (int r = 0; r < 8; ++r) {
    int m = mb + r;
    int node = m / 3, i = m - node * 3;
    out[(size_t)node * 320 + 128 + u * 3 + i] =
        COS_A * self1[(size_t)m * 64 + u] + sc * acc[r];
  }
}

// ---------------- host ----------------
extern "C" void kernel_launch(void* const* d_in, const int* in_sizes, int n_in,
                              void* d_out, int out_size, void* d_ws,
                              size_t ws_size, hipStream_t stream) {
  (void)in_sizes; (void)n_in; (void)out_size; (void)ws_size;
  const float* node_input = (const float*)d_in[0];
  const float* edge_attr  = (const float*)d_in[1];
  const float* esa        = (const float*)d_in[2];
  const float* Wa0 = (const float*)d_in[3];
  const float* Wa1 = (const float*)d_in[4];
  const float* Wb0 = (const float*)d_in[5];
  const float* Wb1 = (const float*)d_in[6];
  const float* M1  = (const float*)d_in[7];
  const float* M2  = (const float*)d_in[8];
  const float* Wtp0 = (const float*)d_in[9];
  const float* Wtp1 = (const float*)d_in[10];
  const float* Wtp2 = (const float*)d_in[11];
  const float* Wtp3 = (const float*)d_in[12];
  const float* Wo0 = (const float*)d_in[13];
  const float* Wo1 = (const float*)d_in[14];
  const int* edge_src = (const int*)d_in[15];
  const int* edge_dst = (const int*)d_in[16];
  float* out = (float*)d_out;

  // workspace carving (~78.3 MB total)
  char* ws = (char*)d_ws;
  size_t off = 0;
  auto carve = [&](size_t bytes) {
    void* p = ws + off;
    off = (off + bytes + 255) & ~(size_t)255;
    return p;
  };
  float* feat0 = (float*)carve((size_t)N_NODES * 128 * 4);
  float* self0 = (float*)carve((size_t)N_NODES * 128 * 4);
  float* feat1 = (float*)carve((size_t)N_NODES * 3 * 64 * 4);
  float* self1 = (float*)carve((size_t)N_NODES * 3 * 64 * 4);
  float* agg   = (float*)carve((size_t)N_NODES * 768 * 4);
  _Float16* xa0 = (_Float16*)carve((size_t)N_NODES * 128 * 2);
  _Float16* xa1 = (_Float16*)carve((size_t)N_NODES * 3 * 64 * 2);
  _Float16* za0 = (_Float16*)carve((size_t)N_NODES * 192 * 2);
  _Float16* za1 = (_Float16*)carve((size_t)N_NODES * 3 * 192 * 2);
  _Float16* Wa0T  = (_Float16*)carve(128 * 128 * 2);
  _Float16* Wb0T  = (_Float16*)carve(128 * 128 * 2);
  _Float16* Wa1T  = (_Float16*)carve(64 * 64 * 2);
  _Float16* Wb1T  = (_Float16*)carve(64 * 64 * 2);
  _Float16* M2T   = (_Float16*)carve(64 * 64 * 2);
  _Float16* Wtp0T = (_Float16*)carve(64 * 128 * 2);
  _Float16* Wtp1T = (_Float16*)carve(64 * 128 * 2);
  _Float16* Wtp2T = (_Float16*)carve(64 * 64 * 2);
  _Float16* Wtp3T = (_Float16*)carve(64 * 64 * 2);
  _Float16* Wo0T  = (_Float16*)carve(192 * 128 * 2);
  _Float16* Wo1T  = (_Float16*)carve(192 * 64 * 2);

  auto launch_w = [&](const float* W, _Float16* WT, int K, int N) {
    int tot = K * N;
    k_w2h<<<(tot + 255) / 256, 256, 0, stream>>>(W, WT, K, N);
  };
  launch_w(Wa0, Wa0T, 128, 128);
  launch_w(Wb0, Wb0T, 128, 128);
  launch_w(Wa1, Wa1T, 64, 64);
  launch_w(Wb1, Wb1T, 64, 64);
  launch_w(M2, M2T, 64, 64);
  launch_w(Wtp0, Wtp0T, 64, 128);
  launch_w(Wtp1, Wtp1T, 64, 128);
  launch_w(Wtp2, Wtp2T, 64, 64);
  launch_w(Wtp3, Wtp3T, 64, 64);
  launch_w(Wo0, Wo0T, 192, 128);
  launch_w(Wo1, Wo1T, 192, 64);

  k_x<<<(N_NODES * 320 + 255) / 256, 256, 0, stream>>>(node_input, xa0, xa1);
  k_zero<<<(N_NODES * 768 + 255) / 256, 256, 0, stream>>>(agg, N_NODES * 768);

  // node GEMMs: feat0/self0 (625 m-tiles x 8 n-tiles), feat1/self1 (1875 x 4)
  k_gemm_dual<<<625, 256, 0, stream>>>(xa0, Wa0T, Wb0T, feat0, self0, 625, 8,
                                       128, 128, INV128);
  k_gemm_dual<<<(1875 * 4 + 7) / 8, 256, 0, stream>>>(xa1, Wa1T, Wb1T, feat1,
                                                      self1, 1875, 4, 64, 64,
                                                      INV64S);

  // fused edge MLP + tensor product + atomic segment-sum: 10000 waves
  k_edge<<<1250, 256, 0, stream>>>(esa, edge_attr, edge_src, edge_dst, M1, M2T,
                                   Wtp0T, Wtp1T, Wtp2T, Wtp3T, feat0, feat1,
                                   agg);

  k_repack<<<(N_NODES * 768 + 255) / 256, 256, 0, stream>>>(agg, za0, za1);

  k_out0<<<625, 256, 0, stream>>>(za0, Wo0T, self0, out);
  k_out1<<<(1875 * 4 + 7) / 8, 256, 0, stream>>>(za1, Wo1T, self1, out);
}
